// dBA_Loss_36197984371246
// MI455X (gfx1250) — matrix operand order repaired and checked by reading, not compiled, not run on hardware
//
#include <hip/hip_runtime.h>
#include <hip/hip_bf16.h>

// ---------------- problem constants ----------------
#define T_LEN     262144
#define BATCH     64
#define NFFT      512
#define STEP      171
#define NSEG      1531          // (262144-512)/171 + 1
#define MTILES    96            // ceil(1531/16)
#define FS_F      3152.0f
#define WIN_POWER 203.4688f     // 512*(0.54^2 + 0.5*0.46^2), periodic Hamming
#define CHUNK     3077          // 15*171 + 512 floats per 16-segment tile
#define TWO_PI    6.28318530717958647692f

typedef __attribute__((ext_vector_type(16))) _Float16 v16h;
typedef __attribute__((ext_vector_type(8)))  _Float16 h8;
typedef __attribute__((ext_vector_type(8)))  float    v8f;

// ---------------------------------------------------------------------------
// Init: f16 DFT table (column-major: Bt[c*512 + k]), window, fused per-column
// weights, and zeroed per-batch band accumulators.
// Column c: bin f = c>>1; even col = cos(2*pi*f*k/512), odd col = sin(...).
// (Sign of imag part is irrelevant: it is squared.)
// ---------------------------------------------------------------------------
__global__ void dba_init(_Float16* __restrict__ Bt, float* __restrict__ win,
                         float* __restrict__ cw, float* __restrict__ band) {
  const int idx = blockIdx.x * blockDim.x + threadIdx.x;
  if (idx < NFFT * NFFT) {
    const int c = idx >> 9;
    const int k = idx & 511;
    const int f = c >> 1;
    const float ang = TWO_PI * (float)f * (float)k * (1.0f / 512.0f);
    const float v = (c & 1) ? __sinf(ang) : __cosf(ang);
    Bt[idx] = (_Float16)v;
  }
  if (idx < NFFT) {
    // periodic Hamming
    win[idx] = 0.54f - 0.46f * __cosf(TWO_PI * (float)idx * (1.0f / 512.0f));
    // fused weight: one-sided scale (2x) * A-weight(linear) / (winPower*FS*nseg)
    const int f = idx >> 1;
    float v = 0.0f;
    if (f >= 1 && f < 244) {               // band [lo=0, hi=244); aw(0)=0 anyway
      const float fr  = (float)f * (FS_F / 512.0f);
      const float fsq = fr * fr;
      const float f1 = 20.6f, f2 = 107.7f, f3 = 737.9f, f4 = 12194.0f;
      const float num = (f4 * f4) * fsq * fsq;
      const float den = (fsq + f1 * f1) * sqrtf(fsq + f2 * f2) *
                        sqrtf(fsq + f3 * f3) * (fsq + f4 * f4);
      const float A = 20.0f * log10f(num / den) + 2.0f;   // -A1000, A1000=-2
      const float aw = powf(10.0f, 0.1f * A);
      v = 2.0f * aw / (WIN_POWER * FS_F * (float)NSEG);
    }
    cw[idx] = v;
  }
  if (idx < BATCH) band[idx] = 0.0f;
}

// ---------------------------------------------------------------------------
// Fused DFT-GEMM + |.|^2 + weighted band reduction.
// Block = 16 segments of one batch (M-tile), 256 threads = 8 waves.
// Wave w computes N-tiles {w, w+8, w+16, w+24} (16 cols each, 512 cols total
// = 256 bins, weights zero beyond bin 243). K = 512 in 16 WMMA steps, with
// the 4 B fragments software-pipelined one step ahead so each WMMA's operand
// loads have a full step (4 WMMAs + A build) of latency cover.
// ---------------------------------------------------------------------------
__global__ __launch_bounds__(256) void dba_gemm(
    const float* __restrict__ x, const _Float16* __restrict__ Bt,
    const float* __restrict__ win, const float* __restrict__ cw,
    float* __restrict__ band) {
  __shared__ float xs[CHUNK];
  __shared__ __align__(32) _Float16 Aw[16 * NFFT];   // windowed f16 A panel
  __shared__ float red[8];

  const int tid = threadIdx.x;
  const int b   = blockIdx.x / MTILES;
  const int mt  = blockIdx.x % MTILES;

  // --- stage x chunk (overlapping span of 16 segments) into LDS ---
  const long base = (long)b * T_LEN + (long)mt * (16 * STEP);
  const int  lim  = T_LEN - mt * (16 * STEP);   // valid tail length
  for (int i = tid; i < CHUNK; i += 256)
    xs[i] = (i < lim) ? x[base + i] : 0.0f;
  __syncthreads();

  // --- build windowed f16 A panel; zero out-of-range segments ---
  for (int e = tid; e < 16 * NFFT; e += 256) {
    const int m = e >> 9;
    const int k = e & 511;
    const int s = mt * 16 + m;
    const float v = (s < NSEG) ? xs[m * STEP + k] * win[k] : 0.0f;
    Aw[e] = (_Float16)v;
  }
  __syncthreads();

  const int w    = tid >> 5;
  const int lane = tid & 31;
  const int m    = lane & 15;   // A row within tile
  const int n    = lane & 15;   // B column within N-tile
  const int lh   = lane >> 4;   // lane half

  // A fragment source: lane holds 16 contiguous K values (k0 + 16*lh + 0..15).
  const h8* Ap = (const h8*)(Aw + m * NFFT + lh * 16);
  // B fragment source (column-major table): lane holds 16 contiguous K values.
  const _Float16* B0 = Bt + ((w     ) * 16 + n) * NFFT + lh * 16;
  const _Float16* B1 = Bt + ((w +  8) * 16 + n) * NFFT + lh * 16;
  const _Float16* B2 = Bt + ((w + 16) * 16 + n) * NFFT + lh * 16;
  const _Float16* B3 = Bt + ((w + 24) * 16 + n) * NFFT + lh * 16;

  v8f acc0 = {}, acc1 = {}, acc2 = {}, acc3 = {};

  // prologue: load step-0 B fragments
  v16h b0 = *(const v16h*)(B0);
  v16h b1 = *(const v16h*)(B1);
  v16h b2 = *(const v16h*)(B2);
  v16h b3 = *(const v16h*)(B3);

#pragma unroll 4
  for (int k0 = 0; k0 < NFFT; k0 += 32) {
    // prefetch next step's B fragments (wrap keeps last prefetch in-bounds;
    // its result is discarded after the loop).
    const int kn = (k0 + 32) & (NFFT - 1);
    const v16h nb0 = *(const v16h*)(B0 + kn);
    const v16h nb1 = *(const v16h*)(B1 + kn);
    const v16h nb2 = *(const v16h*)(B2 + kn);
    const v16h nb3 = *(const v16h*)(B3 + kn);

    const h8 alo = Ap[0];
    const h8 ahi = Ap[1];          // +8 halfs
    Ap += 4;                       // advance 32 halfs
    const v16h a = __builtin_shufflevector(alo, ahi,
        0, 1, 2, 3, 4, 5, 6, 7, 8, 9, 10, 11, 12, 13, 14, 15);

    acc0 = __builtin_amdgcn_wmma_f32_16x16x32_f16(false, a, false, b0,
                                                  (short)0, acc0, false, false);
    acc1 = __builtin_amdgcn_wmma_f32_16x16x32_f16(false, a, false, b1,
                                                  (short)0, acc1, false, false);
    acc2 = __builtin_amdgcn_wmma_f32_16x16x32_f16(false, a, false, b2,
                                                  (short)0, acc2, false, false);
    acc3 = __builtin_amdgcn_wmma_f32_16x16x32_f16(false, a, false, b3,
                                                  (short)0, acc3, false, false);
    b0 = nb0; b1 = nb1; b2 = nb2; b3 = nb3;
  }

  // --- fused epilogue: sum cw[c] * (re^2 + im^2) over this block's tile ---
  const float w0 = cw[(w     ) * 16 + n];
  const float w1 = cw[(w +  8) * 16 + n];
  const float w2 = cw[(w + 16) * 16 + n];
  const float w3 = cw[(w + 24) * 16 + n];
  float partial = 0.0f;
#pragma unroll
  for (int j = 0; j < 8; ++j) {
    partial += acc0[j] * acc0[j] * w0;
    partial += acc1[j] * acc1[j] * w1;
    partial += acc2[j] * acc2[j] * w2;
    partial += acc3[j] * acc3[j] * w3;
  }
#pragma unroll
  for (int off = 16; off > 0; off >>= 1)
    partial += __shfl_xor(partial, off, 32);
  if (lane == 0) red[w] = partial;
  __syncthreads();
  if (tid == 0) {
    float s = 0.0f;
#pragma unroll
    for (int i = 0; i < 8; ++i) s += red[i];
    atomicAdd(&band[b], s);
  }
}

// ---------------------------------------------------------------------------
// Finalize: mean over batch of 10*log10(band[b]).
// ---------------------------------------------------------------------------
__global__ void dba_final(const float* __restrict__ band,
                          float* __restrict__ out) {
  __shared__ float sh[BATCH];
  const int t = threadIdx.x;
  sh[t] = 10.0f * log10f(band[t]);
  __syncthreads();
  if (t == 0) {
    float s = 0.0f;
#pragma unroll
    for (int i = 0; i < BATCH; ++i) s += sh[i];
    out[0] = s * (1.0f / (float)BATCH);
  }
}

extern "C" void kernel_launch(void* const* d_in, const int* in_sizes, int n_in,
                              void* d_out, int out_size, void* d_ws,
                              size_t ws_size, hipStream_t stream) {
  (void)in_sizes; (void)n_in; (void)out_size; (void)ws_size;
  const float* x = (const float*)d_in[0];

  // workspace layout: Bt (512KB f16) | win (2KB) | cw (2KB) | band (256B)
  char* ws = (char*)d_ws;
  _Float16* Bt  = (_Float16*)ws;
  float*    win = (float*)(ws + 512 * 1024);
  float*    cw  = (float*)(ws + 512 * 1024 + 2048);
  float*    bnd = (float*)(ws + 512 * 1024 + 4096);

  dba_init<<<(NFFT * NFFT + 255) / 256, 256, 0, stream>>>(Bt, win, cw, bnd);
  dba_gemm<<<BATCH * MTILES, 256, 0, stream>>>(x, Bt, win, cw, bnd);
  dba_final<<<1, BATCH, 0, stream>>>(bnd, (float*)d_out);
}
